// PagedAttentionLayer_59158879535571
// MI455X (gfx1250) — compile-verified, matrix operand order checked by reading, not run on hardware
//
#include <hip/hip_runtime.h>
#include <hip/hip_bf16.h>

typedef __attribute__((ext_vector_type(16))) _Float16     v16h;
typedef __attribute__((ext_vector_type(8)))  float        v8f;
typedef __attribute__((ext_vector_type(4)))  unsigned int v4u;
typedef __attribute__((ext_vector_type(8)))  int          v8i;
typedef __attribute__((ext_vector_type(4)))  int          v4i;

#define BQ     16
#define DMODEL 4096
#define NH     32
#define NKV    8
#define DK     128
#define QKV_LD 6144   // [Q(4096) | K(1024) | V(1024)]
#define MAXLEN 4096

// Load a 16-bit WMMA fragment slice for one lane from a contiguous f32 row.
// Per the CDNA5 16-bit 16x32 layout, a lane holds k = {0..7} and {16..23}
// relative to its (row-base + half*8) pointer. Works for global or LDS rows.
__device__ __forceinline__ v16h load_frag16(const float* p) {
  float4 f0 = *(const float4*)(p + 0);
  float4 f1 = *(const float4*)(p + 4);
  float4 f2 = *(const float4*)(p + 16);
  float4 f3 = *(const float4*)(p + 20);
  v16h r;
  r[0]  = (_Float16)f0.x; r[1]  = (_Float16)f0.y; r[2]  = (_Float16)f0.z; r[3]  = (_Float16)f0.w;
  r[4]  = (_Float16)f1.x; r[5]  = (_Float16)f1.y; r[6]  = (_Float16)f1.z; r[7]  = (_Float16)f1.w;
  r[8]  = (_Float16)f2.x; r[9]  = (_Float16)f2.y; r[10] = (_Float16)f2.z; r[11] = (_Float16)f2.w;
  r[12] = (_Float16)f3.x; r[13] = (_Float16)f3.y; r[14] = (_Float16)f3.z; r[15] = (_Float16)f3.w;
  return r;
}

// ---- Tensor Data Mover: 2D f32 tile (tile_d1 rows x tile_d0 elems, row
// stride stride_elems) from global memory into LDS at byte offset lds_off.
// D# packing per CDNA5 ISA. All descriptor
// words must be wave-uniform (SGPRs) -> readfirstlane.
__device__ __forceinline__ void tdm_load_tile(unsigned lds_off,
                                              unsigned long long gaddr,
                                              unsigned tile_d0, unsigned tile_d1,
                                              unsigned long long stride_elems) {
  unsigned galo = __builtin_amdgcn_readfirstlane((unsigned)(gaddr & 0xffffffffull));
  unsigned gahi = __builtin_amdgcn_readfirstlane((unsigned)(gaddr >> 32));
  unsigned loff = __builtin_amdgcn_readfirstlane(lds_off);
  v4u g0;
  g0[0] = 1u;                                    // count=1, user descriptor
  g0[1] = loff;                                  // lds_addr [63:32]
  g0[2] = galo;                                  // global_addr [95:64]
  g0[3] = (gahi & 0x1ffffffu) | (2u << 30);      // global_addr[56:32] | type=2
  v8i g1;
  g1[0] = (int)(2u << 16);                                                   // data_size=4B
  g1[1] = (int)((tile_d0 & 0xffffu) << 16);                                  // tensor_dim0[15:0]
  g1[2] = (int)(((tile_d0 >> 16) & 0xffffu) | ((tile_d1 & 0xffffu) << 16));  // dim0 hi | dim1 lo
  g1[3] = (int)(((tile_d1 >> 16) & 0xffffu) | ((tile_d0 & 0xffffu) << 16));  // dim1 hi | tile_dim0
  g1[4] = (int)(tile_d1 & 0xffffu);                                          // tile_dim1, tile_dim2=0
  g1[5] = (int)(stride_elems & 0xffffffffull);                               // dim0_stride lo
  g1[6] = (int)((stride_elems >> 32) & 0xffffull);                           // dim0_stride hi
  g1[7] = 0;
  v4i z4 = {0, 0, 0, 0};
#if __has_include(<hip/amd_detail/amd_gfx1250_TDM.h>)
  v8i z8 = {0, 0, 0, 0, 0, 0, 0, 0};
  __builtin_amdgcn_tensor_load_to_lds(g0, g1, z4, z4, z8, 0);
#else
  __builtin_amdgcn_tensor_load_to_lds(g0, g1, z4, z4, 0);
#endif
}

// Y[0:16, colOff + n0 : colOff + n0 + 16] = X(16xK) @ W(NxK)^T, one wave per tile.
__global__ void gemm_xwt_kernel(const float* __restrict__ X,
                                const float* __restrict__ W,
                                float* __restrict__ Y,
                                int K, int Ntiles, int ldY, int colOff) {
  int wave = threadIdx.x >> 5;
  int lane = threadIdx.x & 31;
  int half = lane >> 4;
  int lr   = lane & 15;
  int tile = blockIdx.x * (blockDim.x >> 5) + wave;
  if (tile >= Ntiles) return;
  int n0 = tile * 16;

  const float* xrow = X + (size_t)lr * K + half * 8;            // A row = batch lr
  const float* wrow = W + (size_t)(n0 + lr) * K + half * 8;     // B col = feature lr

  v8f acc = {};
  for (int k0 = 0; k0 < K; k0 += 32) {
    v16h a = load_frag16(xrow + k0);
    v16h b = load_frag16(wrow + k0);
    __builtin_prefetch(wrow + k0 + 64, 0, 0);
    acc = __builtin_amdgcn_wmma_f32_16x16x32_f16(false, a, false, b,
                                                 (short)0, acc, false, false);
  }
  #pragma unroll
  for (int i = 0; i < 8; ++i) {
    int m = i + 8 * half;
    Y[(size_t)m * ldY + colOff + n0 + lr] = acc[i];
  }
}

__global__ void rope_kernel(const float* __restrict__ qkv,
                            const int* __restrict__ positions,
                            float* __restrict__ q_r,
                            float* __restrict__ k_r) {
  int idx = blockIdx.x * blockDim.x + threadIdx.x;
  const int totalQ = BQ * NH * 64;
  const int total  = totalQ + BQ * NKV * 64;
  if (idx >= total) return;
  const float* src; float* dst; int b, d;
  if (idx < totalQ) {
    b = idx / (NH * 64); int r = idx % (NH * 64); int h = r / 64; d = r % 64;
    src = qkv + (size_t)b * QKV_LD + h * DK;
    dst = q_r + ((size_t)b * NH + h) * DK;
  } else {
    int j = idx - totalQ;
    b = j / (NKV * 64); int r = j % (NKV * 64); int h = r / 64; d = r % 64;
    src = qkv + (size_t)b * QKV_LD + DMODEL + h * DK;
    dst = k_r + ((size_t)b * NKV + h) * DK;
  }
  float inv = __powf(10000.0f, -(float)d * (1.0f / 64.0f));
  float ang = (float)positions[b] * inv;
  float cs = __cosf(ang), sn = __sinf(ang);
  float x0 = src[d], x1 = src[d + 64];
  dst[d]      = x0 * cs - x1 * sn;
  dst[d + 64] = x1 * cs + x0 * sn;
}

// One block per (b, kv head). 128 threads = 4 waves.
__global__ void paged_attn_kernel(const float* __restrict__ qkv,
                                  const float* __restrict__ q_r,
                                  const float* __restrict__ k_r,
                                  const float* __restrict__ k_pool,
                                  const float* __restrict__ v_pool,
                                  const int* __restrict__ positions,
                                  const int* __restrict__ block_table,
                                  float* __restrict__ attn_out) {
  extern __shared__ float smem[];
  float* s_scores = smem;                 // [4][MAXLEN]
  float* s_stat   = smem + 4 * MAXLEN;    // m[0..3], l[0..3]
  float* s_out    = s_stat + 8;           // [4][DK]
  float* s_kv     = s_out + 4 * DK;       // 4 waves x 2 buffers x 4096 floats

  int b    = blockIdx.x >> 3;
  int kv   = blockIdx.x & 7;
  int tid  = threadIdx.x;
  int wave = tid >> 5;
  int lane = tid & 31;
  int half = lane >> 4;
  int lr   = lane & 15;

  int pos = positions[b];
  int L = pos + 1;
  int ntiles = (L + 15) >> 4;
  int nchunk = (L + 31) >> 5;
  int lim = ntiles * 16;
  const float scale = 0.08838834764831845f; // 1/sqrt(128)
  const int* btab = block_table + b * 256;
  float* stage = s_kv + wave * 8192;        // this wave's double buffer

  // Q fragments (rows 0..3 = grouped heads, rows 4..15 zero), reused all tiles.
  v16h aq[4] = {};
  if (lr < 4) {
    const float* qrow = q_r + ((size_t)b * NH + kv * 4 + lr) * DK;
    #pragma unroll
    for (int dc = 0; dc < 4; ++dc)
      aq[dc] = load_frag16(qrow + dc * 32 + half * 8);
  }

  // ---- Phase 1: scores = Q @ K^T. K tiles streamed to LDS by the TDM,
  // double buffered on TENSORcnt; WMMA over D in 4 chunks of 32.
  if (wave < ntiles) {
    int blk = btab[wave];
    const float* g = k_pool + ((size_t)(blk << 4) * NKV + kv) * DK;
    tdm_load_tile((unsigned)(size_t)stage, (unsigned long long)(size_t)g,
                  DK, 16, NKV * DK);
  }
  int kb = 0;
  for (int t = wave; t < ntiles; t += 4, kb ^= 1) {
    if (t + 4 < ntiles) {
      int blk = btab[t + 4];
      const float* g = k_pool + ((size_t)(blk << 4) * NKV + kv) * DK;
      tdm_load_tile((unsigned)(size_t)(stage + (kb ^ 1) * 4096),
                    (unsigned long long)(size_t)g, DK, 16, NKV * DK);
      __builtin_amdgcn_s_wait_tensorcnt((short)1);   // current tile landed
    } else {
      __builtin_amdgcn_s_wait_tensorcnt((short)0);
    }
    float* kt = stage + kb * 4096;
    if ((pos >> 4) == t) {                 // overwrite with roped new K (scatter)
      const float* kr = k_r + ((size_t)b * NKV + kv) * DK;
      float* row = kt + (pos & 15) * DK;
      #pragma unroll
      for (int j = 0; j < 4; ++j) row[lane * 4 + j] = kr[lane * 4 + j];
    }
    int tok = t * 16 + lr;
    const float* kp = kt + lr * DK + half * 8;   // B col = token lr, from LDS
    v8f c = {};
    #pragma unroll
    for (int dc = 0; dc < 4; ++dc) {
      v16h bk = load_frag16(kp + dc * 32);
      c = __builtin_amdgcn_wmma_f32_16x16x32_f16(false, aq[dc], false, bk,
                                                 (short)0, c, false, false);
    }
    if (half == 0) {
      bool valid = tok < L;
      #pragma unroll
      for (int i = 0; i < 4; ++i)
        s_scores[i * MAXLEN + tok] = valid ? c[i] * scale : -1.0e30f;
    }
  }
  __syncthreads();

  // ---- Phase 2: per-row global max and sum-of-exp (wave w owns row w)
  {
    int row = wave;
    float m = -1.0e30f;
    for (int i = lane; i < lim; i += 32) m = fmaxf(m, s_scores[row * MAXLEN + i]);
    #pragma unroll
    for (int off = 16; off >= 1; off >>= 1) m = fmaxf(m, __shfl_xor(m, off));
    float sum = 0.0f;
    for (int i = lane; i < lim; i += 32) sum += __expf(s_scores[row * MAXLEN + i] - m);
    #pragma unroll
    for (int off = 16; off >= 1; off >>= 1) sum += __shfl_xor(sum, off);
    if (lane == 0) { s_stat[row] = m; s_stat[4 + row] = sum; }
  }
  for (int i = tid; i < 4 * DK; i += blockDim.x) s_out[i] = 0.0f;
  __syncthreads();

  // ---- Phase 3: O += P @ V over 32-token chunks. V tile pairs streamed by
  // TDM (two 16-row blocks per chunk), double buffered on TENSORcnt.
  if (wave < nchunk) {
    int base2 = wave * 2;
    const float* g0 = v_pool + ((size_t)(btab[base2] << 4) * NKV + kv) * DK;
    const float* g1 = v_pool + ((size_t)(btab[base2 + 1] << 4) * NKV + kv) * DK;
    tdm_load_tile((unsigned)(size_t)stage, (unsigned long long)(size_t)g0,
                  DK, 16, NKV * DK);
    tdm_load_tile((unsigned)(size_t)stage + 16 * DK * 4,
                  (unsigned long long)(size_t)g1, DK, 16, NKV * DK);
  }
  v8f acc[8] = {};
  int vb = 0;
  for (int ch = wave; ch < nchunk; ch += 4, vb ^= 1) {
    if (ch + 4 < nchunk) {
      int base2 = (ch + 4) * 2;
      const float* g0 = v_pool + ((size_t)(btab[base2] << 4) * NKV + kv) * DK;
      const float* g1 = v_pool + ((size_t)(btab[base2 + 1] << 4) * NKV + kv) * DK;
      unsigned loff = (unsigned)(size_t)(stage + (vb ^ 1) * 4096);
      tdm_load_tile(loff, (unsigned long long)(size_t)g0, DK, 16, NKV * DK);
      tdm_load_tile(loff + 16 * DK * 4, (unsigned long long)(size_t)g1, DK, 16, NKV * DK);
      __builtin_amdgcn_s_wait_tensorcnt((short)2);   // current pair landed
    } else {
      __builtin_amdgcn_s_wait_tensorcnt((short)0);
    }
    float* vt = stage + vb * 4096;
    int base = ch * 32;
    if ((pos >> 5) == ch) {                // overwrite with new V (scatter)
      const float* vr = qkv + (size_t)b * QKV_LD + DMODEL + NKV * DK + kv * DK;
      float* row = vt + (pos - base) * DK;
      #pragma unroll
      for (int j = 0; j < 4; ++j) row[lane * 4 + j] = vr[lane * 4 + j];
    }
    v16h ap = {};
    if (lr < 4) {
      float mrow = s_stat[lr];
      #pragma unroll
      for (int j = 0; j < 8; ++j) {
        int t0 = base + half * 8 + j;
        int t1 = base + 16 + half * 8 + j;
        float p0 = (t0 < lim) ? __expf(s_scores[lr * MAXLEN + t0] - mrow) : 0.0f;
        float p1 = (t1 < lim) ? __expf(s_scores[lr * MAXLEN + t1] - mrow) : 0.0f;
        ap[j]     = (_Float16)p0;
        ap[8 + j] = (_Float16)p1;
      }
    }
    #pragma unroll
    for (int dt = 0; dt < 8; ++dt) {
      int d = dt * 16 + lr;
      v16h bv = {};
      #pragma unroll
      for (int j = 0; j < 8; ++j) {
        bv[j]     = (_Float16)vt[(half * 8 + j) * DK + d];
        bv[8 + j] = (_Float16)vt[(16 + half * 8 + j) * DK + d];
      }
      acc[dt] = __builtin_amdgcn_wmma_f32_16x16x32_f16(false, ap, false, bv,
                                                       (short)0, acc[dt], false, false);
    }
  }
  if (half == 0) {
    #pragma unroll
    for (int dt = 0; dt < 8; ++dt)
      #pragma unroll
      for (int i = 0; i < 4; ++i)
        atomicAdd(&s_out[i * DK + dt * 16 + lr], acc[dt][i]);
  }
  __syncthreads();

  // ---- Phase 4: normalize and write [b][kv*4+g][d]
  float* outp = attn_out + ((size_t)b * NH + kv * 4) * DK;
  for (int i = tid; i < 4 * DK; i += blockDim.x) {
    int g = i >> 7;
    outp[i] = s_out[i] / s_stat[4 + g];
  }
}

extern "C" void kernel_launch(void* const* d_in, const int* in_sizes, int n_in,
                              void* d_out, int out_size, void* d_ws, size_t ws_size,
                              hipStream_t stream) {
  const float* x      = (const float*)d_in[0];
  const float* W_Q    = (const float*)d_in[1];
  const float* W_K    = (const float*)d_in[2];
  const float* W_V    = (const float*)d_in[3];
  const float* W_O    = (const float*)d_in[4];
  const float* k_pool = (const float*)d_in[5];
  const float* v_pool = (const float*)d_in[6];
  const int* positions = (const int*)d_in[7];
  const int* btab      = (const int*)d_in[8];
  float* out = (float*)d_out;

  float* ws  = (float*)d_ws;
  float* qkv = ws;                       // 16*6144
  float* q_r = qkv + 16 * QKV_LD;        // 16*4096 (roped Q)
  float* k_r = q_r + 16 * DMODEL;        // 16*1024 (roped new K)
  float* a_o = k_r + 16 * NKV * DK;      // 16*4096 (attention output)

  dim3 blk(128);
  gemm_xwt_kernel<<<64, blk, 0, stream>>>(x, W_Q, qkv, DMODEL, 256, QKV_LD, 0);
  gemm_xwt_kernel<<<16, blk, 0, stream>>>(x, W_K, qkv, DMODEL, 64, QKV_LD, DMODEL);
  gemm_xwt_kernel<<<16, blk, 0, stream>>>(x, W_V, qkv, DMODEL, 64, QKV_LD, DMODEL + NKV * DK);

  rope_kernel<<<160, 256, 0, stream>>>(qkv, positions, q_r, k_r);

  size_t shmem = (size_t)(4 * MAXLEN + 8 + 4 * DK + 4 * 8192) * sizeof(float);
  paged_attn_kernel<<<BQ * NKV, 128, shmem, stream>>>(qkv, q_r, k_r, k_pool, v_pool,
                                                      positions, btab, a_o);

  gemm_xwt_kernel<<<64, blk, 0, stream>>>(a_o, W_O, out, DMODEL, 256, DMODEL, 0);
}
